// DynamicCNN_50096498540925
// MI455X (gfx1250) — matrix operand order repaired and checked by reading, not compiled, not run on hardware
//
#include <hip/hip_runtime.h>
#include <math.h>

// ---------------------------------------------------------------------------
// Types for CDNA5 WMMA (wave32): bf16 A/B fragments (16 x bf16 = 8 VGPRs),
// f32 C/D (8 VGPRs).
// ---------------------------------------------------------------------------
typedef __attribute__((ext_vector_type(16))) __bf16        v16bf;
typedef __attribute__((ext_vector_type(8)))  float         v8f;
typedef __attribute__((ext_vector_type(8)))  unsigned short us8;

#define LDT 40  // ushort stride of LDS tile rows (80 B => every row 16B aligned)

__device__ __forceinline__ unsigned short f2bf(float f) {
  unsigned int u = __float_as_uint(f);
  u += 0x7FFFu + ((u >> 16) & 1u);   // round-to-nearest-even
  return (unsigned short)(u >> 16);
}

union FragU { us8 u[2]; v16bf v; };

// A fragment, 16x32 bf16: lane m(0..15) holds K=klo..klo+7 (v0..3) and
// K=klo+16..klo+23 (v4..7) with klo = 8*(lane/16).  As is row-major [row][k].
__device__ __forceinline__ v16bf load_a_frag(const unsigned short* As, int row, int lane) {
  const unsigned short* p = As + row * LDT + ((lane >> 4) << 3);
  FragU f; f.u[0] = *(const us8*)p; f.u[1] = *(const us8*)(p + 16);
  return f.v;
}
// B fragment, 32x16 bf16: lane holds col = lane%16, K = klo..klo+7 (v0..3) and
// K=klo+16..klo+23 (v4..7) with klo = 8*(lane/16).  Bs is stored transposed: [col][k].
__device__ __forceinline__ v16bf load_b_frag(const unsigned short* Bs, int col, int lane) {
  const unsigned short* p = Bs + col * LDT + ((lane >> 4) << 3);
  FragU f; f.u[0] = *(const us8*)p; f.u[1] = *(const us8*)(p + 16);
  return f.v;
}

// ---------------------------------------------------------------------------
// Generic GEMM:  C[M,N] = act(A[M,K] @ B[K,N] + bias[N])
// fp32 in memory, bf16 WMMA math with fp32 accumulation.
// Requires K % 32 == 0 and N % 4 == 0 (true for every call below).
// 256 threads = 8 waves (4 along M x 2 along N); wave tile 32x64 = 8 WMMAs.
// act: 0 = none, 1 = relu, 2 = tanh
// ---------------------------------------------------------------------------
__global__ __launch_bounds__(256) void gemm_bf16_kernel(
    const float* __restrict__ A, const float* __restrict__ B,
    const float* __restrict__ bias, float* __restrict__ C,
    int M, int N, int K, int act)
{
  __shared__ __align__(16) unsigned short As[128 * LDT];
  __shared__ __align__(16) unsigned short Bs[128 * LDT];
  const int tid  = threadIdx.x;
  const int lane = tid & 31, wave = tid >> 5;
  const int wm = wave & 3, wn = wave >> 2;
  const int m0 = blockIdx.y * 128, n0 = blockIdx.x * 128;

  v8f acc[2][4] = {};

  for (int k0 = 0; k0 < K; k0 += 32) {
    // --- A tile: 128 x 32 fp32 -> bf16 LDS (row-major) ---
    for (int i = 0; i < 4; ++i) {
      int q = tid + (i << 8);            // 0..1023 quads
      int r = q >> 3, c4 = (q & 7) << 2; // row 0..127, k 0..28
      int gm = m0 + r;
      float4 f = (gm < M) ? *(const float4*)(A + (size_t)gm * K + k0 + c4)
                          : make_float4(0.f, 0.f, 0.f, 0.f);
      unsigned short* d = As + r * LDT + c4;
      d[0] = f2bf(f.x); d[1] = f2bf(f.y); d[2] = f2bf(f.z); d[3] = f2bf(f.w);
    }
    // --- B tile: 32 x 128 fp32 -> bf16 LDS, transposed to [n][k] ---
    for (int i = 0; i < 4; ++i) {
      int q = tid + (i << 8);
      int r = q >> 5, c4 = (q & 31) << 2; // k 0..31, n 0..124
      int gn = n0 + c4;
      float4 f = (gn < N) ? *(const float4*)(B + (size_t)(k0 + r) * N + gn)
                          : make_float4(0.f, 0.f, 0.f, 0.f);
      Bs[(c4 + 0) * LDT + r] = f2bf(f.x);
      Bs[(c4 + 1) * LDT + r] = f2bf(f.y);
      Bs[(c4 + 2) * LDT + r] = f2bf(f.z);
      Bs[(c4 + 3) * LDT + r] = f2bf(f.w);
    }
    __syncthreads();
    v16bf af[2], bfr[4];
    for (int mi = 0; mi < 2; ++mi)
      af[mi] = load_a_frag(As, wm * 32 + mi * 16 + (lane & 15), lane);
    for (int ni = 0; ni < 4; ++ni)
      bfr[ni] = load_b_frag(Bs, wn * 64 + ni * 16 + (lane & 15), lane);
    for (int mi = 0; mi < 2; ++mi)
      for (int ni = 0; ni < 4; ++ni)
        acc[mi][ni] = __builtin_amdgcn_wmma_f32_16x16x32_bf16(
            false, af[mi], false, bfr[ni], (short)0, acc[mi][ni], false, false);
    __syncthreads();
  }

  // C layout: VGPR r -> M = r (lanes 0-15) / r+8 (lanes 16-31); N = lane%16
  const int rhi = (lane >> 4) << 3;
  for (int mi = 0; mi < 2; ++mi)
    for (int ni = 0; ni < 4; ++ni) {
      int col = n0 + wn * 64 + ni * 16 + (lane & 15);
      if (col >= N) continue;
      float bv = bias ? bias[col] : 0.f;
      for (int r = 0; r < 8; ++r) {
        int row = m0 + wm * 32 + mi * 16 + rhi + r;
        if (row >= M) continue;
        float vv = acc[mi][ni][r] + bv;
        if (act == 1) vv = fmaxf(vv, 0.f);
        else if (act == 2) vv = tanhf(vv);
        C[(size_t)row * N + col] = vv;
      }
    }
}

// ---------------------------------------------------------------------------
// Softmax-pool:  fw = softmax_l(kh + mask), qq[b,c] = sum_l fw * v
// kh, v: (16, 512, 1024). One thread per (b, column c = h*Q+q).
// ---------------------------------------------------------------------------
__global__ __launch_bounds__(256) void pool_kernel(
    const float* __restrict__ kh, const float* __restrict__ v,
    const int* __restrict__ qmask, float* __restrict__ qq)
{
  int b = blockIdx.y;
  int c = blockIdx.x * 256 + threadIdx.x;
  const float* Kp = kh + (size_t)b * 512 * 1024 + c;
  const float* Vp = v  + (size_t)b * 512 * 1024 + c;
  const int*   qm = qmask + b * 512;
  float mx = -3.4e38f;
  for (int l = 0; l < 512; ++l)
    if (qm[l]) mx = fmaxf(mx, Kp[(size_t)l * 1024]);
  float den = 0.f, num = 0.f;
  for (int l = 0; l < 512; ++l)
    if (qm[l]) {
      float w = __expf(Kp[(size_t)l * 1024] - mx);
      den += w;
      num += w * Vp[(size_t)l * 1024];
    }
  qq[b * 1024 + c] = num / den;
}

// ---------------------------------------------------------------------------
// Dynamic conv layer GEMM (per bh):  xout[l,e] = relu(A @ W_bh + b_bh) * am[l]
//   layer 0 : A(l,d) = a_hidden[b,l,d] * am[b,l]            (L=512, din=512)
//   layer>0 : A(l,d) = concat(x[l], x[l+1])  from xin[bh]   (Lprev = L+1)
// W_bh = pvec[bh, 0 : din*e] row-major (din, e); bias at pvec[bh, din*e : +e]
// Requires din % 32 == 0 (holds: 512,256,128,64,32).  e <= 128 -> grid.x = 1.
// ---------------------------------------------------------------------------
__global__ __launch_bounds__(256) void conv_bf16_kernel(
    const float* __restrict__ xin, const float* __restrict__ ah,
    const float* __restrict__ pvec, const int* __restrict__ amask,
    float* __restrict__ xout,
    int L, int e, int din, int eprev, int Np, int layer0)
{
  __shared__ __align__(16) unsigned short As[128 * LDT];
  __shared__ __align__(16) unsigned short Bs[128 * LDT];
  const int tid  = threadIdx.x;
  const int lane = tid & 31, wave = tid >> 5;
  const int wm = wave & 3, wn = wave >> 2;
  const int bh = blockIdx.z, b = bh >> 1;
  const int m0 = blockIdx.y * 128;
  const float* Bmat  = pvec + (size_t)bh * Np;       // (din, e) row-major
  const float* cbias = Bmat + (size_t)din * e;       // e entries

  v8f acc[2][4] = {};

  for (int k0 = 0; k0 < din; k0 += 32) {
    for (int i = 0; i < 4; ++i) {
      int q = tid + (i << 8);
      int r = q >> 3, c4 = (q & 7) << 2;
      int gm = m0 + r;
      unsigned short* d = As + r * LDT + c4;
      for (int j = 0; j < 4; ++j) {
        int gk = k0 + c4 + j;
        float vv = 0.f;
        if (gm < L) {
          if (layer0) {
            vv = ah[((size_t)b * 512 + gm) * 512 + gk] * (float)amask[b * 512 + gm];
          } else {
            int hi = (gk >= eprev);
            int sl = gm + hi;
            int sd = gk - (hi ? eprev : 0);
            vv = xin[(size_t)bh * (L + 1) * eprev + (size_t)sl * eprev + sd];
          }
        }
        d[j] = f2bf(vv);
      }
    }
    for (int i = 0; i < 4; ++i) {
      int q = tid + (i << 8);
      int r = q >> 5, c4 = (q & 31) << 2;
      for (int j = 0; j < 4; ++j) {
        int gn = c4 + j;
        float vv = (gn < e) ? Bmat[(size_t)(k0 + r) * e + gn] : 0.f;
        Bs[gn * LDT + r] = f2bf(vv);
      }
    }
    __syncthreads();
    v16bf af[2], bfr[4];
    for (int mi = 0; mi < 2; ++mi)
      af[mi] = load_a_frag(As, wm * 32 + mi * 16 + (lane & 15), lane);
    for (int ni = 0; ni < 4; ++ni)
      bfr[ni] = load_b_frag(Bs, wn * 64 + ni * 16 + (lane & 15), lane);
    for (int mi = 0; mi < 2; ++mi)
      for (int ni = 0; ni < 4; ++ni)
        acc[mi][ni] = __builtin_amdgcn_wmma_f32_16x16x32_bf16(
            false, af[mi], false, bfr[ni], (short)0, acc[mi][ni], false, false);
    __syncthreads();
  }

  const int rhi = (lane >> 4) << 3;
  for (int mi = 0; mi < 2; ++mi)
    for (int ni = 0; ni < 4; ++ni) {
      int col = wn * 64 + ni * 16 + (lane & 15);
      if (col >= e) continue;
      float cb = cbias[col];
      for (int r = 0; r < 8; ++r) {
        int row = m0 + wm * 32 + mi * 16 + rhi + r;
        if (row >= L) continue;
        float vv = fmaxf(acc[mi][ni][r] + cb, 0.f) * (float)amask[b * 512 + row];
        xout[(size_t)bh * L * e + (size_t)row * e + col] = vv;
      }
    }
}

// ---------------------------------------------------------------------------
// Per-(b,h): layernorm each row of x over e (in place), column-max over L,
// then out_feats[bh] += relu(maxf @ feat_w + feat_b).  One block per bh,
// 8 wave32s; one wave per row, shuffle reductions.
// ---------------------------------------------------------------------------
__global__ __launch_bounds__(256) void ln_feat_kernel(
    float* __restrict__ x, const float* __restrict__ pvec,
    const float* __restrict__ g, const float* __restrict__ bt,
    float* __restrict__ outf, int L, int e, int Np, int first)
{
  __shared__ float pmax[8 * 128];
  __shared__ float mf[128];
  const int bh = blockIdx.x;
  const int wave = threadIdx.x >> 5, lane = threadIdx.x & 31;
  float* xb = x + (size_t)bh * L * e;
  const float inve = 1.0f / (float)e;

  float rmax[4];
  for (int j = 0; j < 4; ++j) rmax[j] = -3.4e38f;

  for (int l = wave; l < L; l += 8) {
    float vals[4];
    float s = 0.f;
    for (int j = 0; j < 4; ++j) {
      int c = lane + 32 * j;
      vals[j] = (c < e) ? xb[(size_t)l * e + c] : 0.f;
      s += vals[j];
    }
    for (int o = 16; o > 0; o >>= 1) s += __shfl_xor(s, o);
    float mu = s * inve;
    float vs = 0.f;
    for (int j = 0; j < 4; ++j) {
      int c = lane + 32 * j;
      if (c < e) { float d = vals[j] - mu; vs += d * d; }
    }
    for (int o = 16; o > 0; o >>= 1) vs += __shfl_xor(vs, o);
    float inv = rsqrtf(vs * inve + 1e-5f);
    for (int j = 0; j < 4; ++j) {
      int c = lane + 32 * j;
      if (c < e) {
        float y = (vals[j] - mu) * inv * g[c] + bt[c];
        xb[(size_t)l * e + c] = y;
        rmax[j] = fmaxf(rmax[j], y);
      }
    }
  }
  for (int j = 0; j < 4; ++j) {
    int c = lane + 32 * j;
    if (c < e) pmax[wave * 128 + c] = rmax[j];
  }
  __syncthreads();
  int t = threadIdx.x;
  if (t < e) {
    float m = pmax[t];
    for (int w = 1; w < 8; ++w) m = fmaxf(m, pmax[w * 128 + t]);
    mf[t] = m;
  }
  __syncthreads();
  if (t < 128) {
    const float* fw = pvec + (size_t)bh * Np + (Np - 128 * (e + 1)); // (e, 128)
    const float* fb = pvec + (size_t)bh * Np + (Np - 128);
    float s = fb[t];
    for (int c = 0; c < e; ++c) s += mf[c] * fw[(size_t)c * 128 + t];
    s = fmaxf(s, 0.f);
    outf[bh * 128 + t] = (first ? 0.f : outf[bh * 128 + t]) + s;
  }
}

// ---------------------------------------------------------------------------
// Classifier head (tiny): single block, 512 threads, all in LDS.
// h1 = bn(relu(a_rep@cW1+cb1)); h2 = bn(relu(h1@cW2+cb2)); out = h2@cW3+cb3
// bn is over the batch dim (16 rows), so each thread owns a full column.
// ---------------------------------------------------------------------------
__global__ __launch_bounds__(512) void head_kernel(
    const float* __restrict__ arep,
    const float* __restrict__ cW1, const float* __restrict__ cb1,
    const float* __restrict__ cg1, const float* __restrict__ cbt1,
    const float* __restrict__ cW2, const float* __restrict__ cb2,
    const float* __restrict__ cg2, const float* __restrict__ cbt2,
    const float* __restrict__ cW3, const float* __restrict__ cb3,
    float* __restrict__ out)
{
  __shared__ float h1[16 * 512];
  __shared__ float h2[16 * 512];
  const int j = threadIdx.x;
  float col[16];

  for (int i = 0; i < 16; ++i) {
    float s = cb1[j];
    for (int k = 0; k < 256; ++k) s += arep[i * 256 + k] * cW1[k * 512 + j];
    col[i] = fmaxf(s, 0.f);
  }
  float mu = 0.f;
  for (int i = 0; i < 16; ++i) mu += col[i];
  mu *= (1.f / 16.f);
  float var = 0.f;
  for (int i = 0; i < 16; ++i) { float d = col[i] - mu; var += d * d; }
  var *= (1.f / 16.f);
  float inv = rsqrtf(var + 1e-5f);
  for (int i = 0; i < 16; ++i) h1[i * 512 + j] = (col[i] - mu) * inv * cg1[j] + cbt1[j];
  __syncthreads();

  for (int i = 0; i < 16; ++i) {
    float s = cb2[j];
    for (int k = 0; k < 512; ++k) s += h1[i * 512 + k] * cW2[k * 512 + j];
    col[i] = fmaxf(s, 0.f);
  }
  mu = 0.f;
  for (int i = 0; i < 16; ++i) mu += col[i];
  mu *= (1.f / 16.f);
  var = 0.f;
  for (int i = 0; i < 16; ++i) { float d = col[i] - mu; var += d * d; }
  var *= (1.f / 16.f);
  inv = rsqrtf(var + 1e-5f);
  for (int i = 0; i < 16; ++i) h2[i * 512 + j] = (col[i] - mu) * inv * cg2[j] + cbt2[j];
  __syncthreads();

  if (j < 64) {
    int i = j >> 2, o = j & 3;
    float s = cb3[o];
    for (int k = 0; k < 512; ++k) s += h2[i * 512 + k] * cW3[k * 4 + o];
    out[i * 4 + o] = s;
  }
}

// ---------------------------------------------------------------------------
// Orchestration
// ---------------------------------------------------------------------------
extern "C" void kernel_launch(void* const* d_in, const int* in_sizes, int n_in,
                              void* d_out, int out_size, void* d_ws, size_t ws_size,
                              hipStream_t stream) {
  (void)in_sizes; (void)n_in; (void)out_size; (void)ws_size;

  const float* qe    = (const float*)d_in[0];   // (16,512,512)
  const int*   qmask = (const int*)  d_in[1];   // (16,512)
  const float* ah    = (const float*)d_in[2];   // (16,512,512)
  const int*   amask = (const int*)  d_in[3];   // (16,512)
  const float* kW1 = (const float*)d_in[4];  const float* kb1 = (const float*)d_in[5];
  const float* kW2 = (const float*)d_in[6];  const float* kb2 = (const float*)d_in[7];
  const float* vW1 = (const float*)d_in[8];  const float* vb1 = (const float*)d_in[9];
  const float* vW2 = (const float*)d_in[10]; const float* vb2 = (const float*)d_in[11];
  const float* gfW[9]; const float* gfb[9]; const float* lng[9]; const float* lnb[9];
  for (int i = 0; i < 9; ++i) {
    gfW[i] = (const float*)d_in[12 + i];
    gfb[i] = (const float*)d_in[21 + i];
    lng[i] = (const float*)d_in[30 + i];
    lnb[i] = (const float*)d_in[39 + i];
  }
  const float* cW1 = (const float*)d_in[48]; const float* cb1 = (const float*)d_in[49];
  const float* cg1 = (const float*)d_in[50]; const float* cbt1= (const float*)d_in[51];
  const float* cW2 = (const float*)d_in[52]; const float* cb2 = (const float*)d_in[53];
  const float* cg2 = (const float*)d_in[54]; const float* cbt2= (const float*)d_in[55];
  const float* cW3 = (const float*)d_in[56]; const float* cb3 = (const float*)d_in[57];

  char* ws = (char*)d_ws;
  float* buf0 = (float*)ws;                               // 8192x2048 = 64 MB
  float* buf1 = (float*)(ws + (size_t)67108864);          // 8192x1024 = 32 MB
  float* buf2 = (float*)(ws + (size_t)100663296);         // 8192x1024 = 32 MB
  float* qq   = (float*)(ws + (size_t)134217728);         // 32x512
  float* outf = qq + 32 * 512;                            // 32x128
  // buf0/1/2 are dead after the pooling stage -> recycle buf0's 64 MB region:
  float* pvec = buf0;                                     // <= 10.6 MB
  float* xA   = (float*)(ws + (size_t)16777216);          // <= 8.4 MB
  float* xB   = (float*)(ws + (size_t)33554432);          // <= 8.4 MB

  const int M = 8192;  // B*Lq
  dim3 blk(256);

  // key path: relu(qe@kW1+kb1) -> tanh(.@kW2+kb2)
  gemm_bf16_kernel<<<dim3(2048 / 128, M / 128), blk, 0, stream>>>(
      qe, kW1, kb1, buf0, M, 2048, 512, 1);
  gemm_bf16_kernel<<<dim3(1024 / 128, M / 128), blk, 0, stream>>>(
      buf0, kW2, kb2, buf1, M, 1024, 2048, 2);
  // value path
  gemm_bf16_kernel<<<dim3(2048 / 128, M / 128), blk, 0, stream>>>(
      qe, vW1, vb1, buf0, M, 2048, 512, 1);
  gemm_bf16_kernel<<<dim3(1024 / 128, M / 128), blk, 0, stream>>>(
      buf0, vW2, vb2, buf2, M, 1024, 2048, 2);
  // softmax pooling -> q_queries (32, 512)
  pool_kernel<<<dim3(4, 16), blk, 0, stream>>>(buf1, buf2, qmask, qq);

  static const int E[9]  = {128, 64, 32, 16, 16, 16, 16, 16, 16};
  static const int IN[9] = {512, 256, 128, 64, 32, 32, 32, 32, 32};

  float* xin = xA;
  float* xout = xB;
  for (int i = 0; i < 9; ++i) {
    const int e = E[i], din = IN[i];
    const int Np = (din + 1) * e + 128 * (e + 1);
    const int L = 512 - i;
    // pvec[bh, Np] = q_queries @ gf_W[i] + gf_b[i]
    gemm_bf16_kernel<<<dim3((Np + 127) / 128, 1), blk, 0, stream>>>(
        qq, gfW[i], gfb[i], pvec, 32, Np, 512, 0);
    // dynamic conv + relu + mask
    conv_bf16_kernel<<<dim3(1, (L + 127) / 128, 32), blk, 0, stream>>>(
        xin, ah, pvec, amask, xout, L, e, din, (i > 0) ? E[i - 1] : 0, Np, i == 0);
    // layernorm (in place) + column max + feature accumulation
    ln_feat_kernel<<<dim3(32), blk, 0, stream>>>(
        xout, pvec, lng[i], lnb[i], outf, L, e, Np, i == 0);
    float* t = xin; xin = xout; xout = t;
  }

  // classifier head -> (16, 4)
  head_kernel<<<dim3(1), dim3(512), 0, stream>>>(
      outf, cW1, cb1, cg1, cbt1, cW2, cb2, cg2, cbt2, cW3, cb3, (float*)d_out);
}
